// GroupedQueryAttention_26852135534954
// MI455X (gfx1250) — compile-verified, matrix operand order checked by reading, not run on hardware
//
#include <hip/hip_runtime.h>
#include <hip/hip_bf16.h>

// ---------------------------------------------------------------------------
// GQA block for MI455X (gfx1250, wave32, WMMA, double-buffered async DMA).
// B=2 S=2048 D_IN=2048 NUM_HEADS=32 NUM_KV_GROUPS=8 HEAD_DIM=64 GROUP=4
// ---------------------------------------------------------------------------

typedef __attribute__((ext_vector_type(16))) __bf16 v16bf;
typedef __attribute__((ext_vector_type(8)))  float  v8f;

union FragAB { uint4 q[2]; v16bf v; };          // 32 bytes = 16 bf16 per lane

__device__ __forceinline__ unsigned short f2bf(float f) {
  unsigned int u = __float_as_uint(f);
  return (unsigned short)((u + 0x7FFFu + ((u >> 16) & 1u)) >> 16);  // RNE
}

__device__ __forceinline__ v8f wmma_bf16(v16bf a, v16bf b, v8f c) {
  // D = A(16x32 bf16) * B(32x16 bf16) + C(16x16 f32)
  return __builtin_amdgcn_wmma_f32_16x16x32_bf16(false, a, false, b,
                                                 (short)0, c, false, false);
}

// Async DMA: 16B global -> LDS per lane, tracked by ASYNCcnt (no VGPR staging).
__device__ __forceinline__ void async_ld_b128(void* lds_ptr, const void* gptr) {
  unsigned lds_off = (unsigned)(size_t)lds_ptr;   // low 32b of flat = LDS byte addr
  asm volatile("global_load_async_to_lds_b128 %0, %1, off"
               :: "v"(lds_off), "v"(gptr) : "memory");
}
// Async loads complete in order per wave: waiting ASYNCcnt<=N with N ops of the
// *next* tile in flight guarantees the *previous* tile's copies have landed.
__device__ __forceinline__ void wait_async0() {
#if __has_builtin(__builtin_amdgcn_s_wait_asynccnt)
  __builtin_amdgcn_s_wait_asynccnt(0);
#else
  asm volatile("s_wait_asynccnt 0" ::: "memory");
#endif
}
__device__ __forceinline__ void wait_async4() {
#if __has_builtin(__builtin_amdgcn_s_wait_asynccnt)
  __builtin_amdgcn_s_wait_asynccnt(4);
#else
  asm volatile("s_wait_asynccnt 4" ::: "memory");
#endif
}

enum : int { BB = 2, SS = 2048, DIN = 2048, NH = 32, NG = 8, HD = 64,
             DOUT = 2048, MALL = BB * SS };

// ---------------------------------------------------------------------------
// fp32 -> bf16 conversion (vector x4)
// ---------------------------------------------------------------------------
__global__ __launch_bounds__(256) void cvt_f32_bf16(
    const float* __restrict__ in, unsigned short* __restrict__ out, int n) {
  int i = (blockIdx.x * 256 + threadIdx.x) * 4;
  if (i < n) {
    float4 f = *(const float4*)(in + i);
    ushort4 o;
    o.x = f2bf(f.x); o.y = f2bf(f.y); o.z = f2bf(f.z); o.w = f2bf(f.w);
    *(ushort4*)(out + i) = o;
  }
}

// ---------------------------------------------------------------------------
// Y[M,N] = A[M,K](bf16) * W[N,K]^T(bf16)
// Block tile 128x128, BK=32, 256 threads = 8 waves (4 M x 2 N), wave = 32x64.
// Double-buffered LDS; tile t+1 DMA overlaps tile t WMMA.
// f32mode: 0 = none, 1 = row-major f32, 2 = kv-transposed f32 ([b,g,s,d]).
// ---------------------------------------------------------------------------
#define LSTR 40   // padded LDS row stride in bf16 elements (80B, 16B aligned)

__global__ __launch_bounds__(256) void gemm_bf16_wmma(
    const unsigned short* __restrict__ A, const unsigned short* __restrict__ W,
    unsigned short* __restrict__ Cb, float* __restrict__ Cf,
    int M, int N, int K, int f32mode) {
  __shared__ unsigned short smA[2][128 * LSTR];
  __shared__ unsigned short smB[2][128 * LSTR];

  const int tid = threadIdx.x;
  const int lane = tid & 31, wave = tid >> 5;
  const int wm = wave >> 1, wn = wave & 1;
  const int bm = blockIdx.y * 128, bn = blockIdx.x * 128;
  const int laneM = lane & 15;
  const int koff0 = (lane < 16) ? 0 : 8;   // ISA 16-bit A/B lane-half K split

  // Per-thread staging coordinates (4 async b128 per thread per tile).
  const int r0 = tid >> 2, seg0 = (tid & 3) * 8;          // idx = tid
  const int r1 = (tid + 256) >> 2, seg1 = seg0;           // idx = tid + 256

  v8f acc[2][4];
#pragma unroll
  for (int mi = 0; mi < 2; ++mi)
#pragma unroll
    for (int ni = 0; ni < 4; ++ni)
#pragma unroll
      for (int e = 0; e < 8; ++e) acc[mi][ni][e] = 0.0f;

  auto stage = [&](int kt, int buf) {
    async_ld_b128(&smA[buf][r0 * LSTR + seg0], A + (size_t)(bm + r0) * K + kt + seg0);
    async_ld_b128(&smB[buf][r0 * LSTR + seg0], W + (size_t)(bn + r0) * K + kt + seg0);
    async_ld_b128(&smA[buf][r1 * LSTR + seg1], A + (size_t)(bm + r1) * K + kt + seg1);
    async_ld_b128(&smB[buf][r1 * LSTR + seg1], W + (size_t)(bn + r1) * K + kt + seg1);
  };

  const int T = K >> 5;          // number of 32-wide K tiles
  stage(0, 0);
  for (int t = 0; t < T; ++t) {
    const int cur = t & 1;
    if (t + 1 < T) {             // prefetch next tile, keep its 4 DMAs in flight
      stage((t + 1) << 5, cur ^ 1);
      wait_async4();             // tile t landed; tile t+1 still streaming
    } else {
      wait_async0();
    }
    __syncthreads();             // tile t visible to all waves

    const unsigned short* sa = smA[cur];
    const unsigned short* sb = smB[cur];
    FragAB af[2], bf[4];
#pragma unroll
    for (int mi = 0; mi < 2; ++mi) {
      int r = wm * 32 + mi * 16 + laneM;
      af[mi].q[0] = *(const uint4*)(&sa[r * LSTR + koff0]);
      af[mi].q[1] = *(const uint4*)(&sa[r * LSTR + koff0 + 16]);
    }
#pragma unroll
    for (int ni = 0; ni < 4; ++ni) {
      int r = wn * 64 + ni * 16 + laneM;
      bf[ni].q[0] = *(const uint4*)(&sb[r * LSTR + koff0]);
      bf[ni].q[1] = *(const uint4*)(&sb[r * LSTR + koff0 + 16]);
    }
#pragma unroll
    for (int mi = 0; mi < 2; ++mi)
#pragma unroll
      for (int ni = 0; ni < 4; ++ni)
        acc[mi][ni] = wmma_bf16(af[mi].v, bf[ni].v, acc[mi][ni]);

    __syncthreads();             // all reads of buf[cur] done before t+2 lands
  }

  const int half8 = (lane >> 4) * 8;
#pragma unroll
  for (int mi = 0; mi < 2; ++mi)
#pragma unroll
    for (int ni = 0; ni < 4; ++ni)
#pragma unroll
      for (int r = 0; r < 8; ++r) {
        int row = bm + wm * 32 + mi * 16 + r + half8;
        int col = bn + wn * 64 + ni * 16 + laneM;
        float v = acc[mi][ni][r];
        if (Cb) Cb[(size_t)row * N + col] = f2bf(v);
        if (f32mode == 1) {
          Cf[(size_t)row * N + col] = v;
        } else if (f32mode == 2) {          // keys/values: [b, g, s, d]
          int b = row >> 11, s = row & 2047;
          int g = col >> 6, d = col & 63;
          Cf[((((size_t)b * NG + g) * SS + s) << 6) + d] = v;
        }
      }
}

// ---------------------------------------------------------------------------
// Causal flash attention, one (b,head) x 128 query rows per workgroup.
// 8 waves x 16 rows.  Q bf16 [B,S,2048], K/V bf16 [B,S,512], ctx bf16 out.
// ---------------------------------------------------------------------------
#define KSTR 72   // 64 + 8 pad, 144B row stride (16B aligned)

__global__ __launch_bounds__(256) void attn_wmma(
    const unsigned short* __restrict__ Qb, const unsigned short* __restrict__ Kb,
    const unsigned short* __restrict__ Vb, unsigned short* __restrict__ Ctx) {
  __shared__ unsigned short smK[64 * KSTR];    // K block, row-major [k][d]
  __shared__ unsigned short smV[64 * KSTR];    // V block, transposed [d][k]
  __shared__ unsigned short smP[128 * KSTR];   // per-wave P staging

  const int tid = threadIdx.x;
  const int lane = tid & 31, wave = tid >> 5;
  const int laneM = lane & 15;
  const int half8 = (lane >> 4) * 8;
  const int koff0 = (lane < 16) ? 0 : 8;

  const int qb0 = blockIdx.x * 128;
  const int bh = blockIdx.y;
  const int b = bh >> 5, h = bh & 31, g = h >> 2;

  // Q fragments: rows qb0 + wave*16 + laneM, 64 d-values in 2 K-chunks.
  FragAB qf[2];
  {
    const unsigned short* qp =
        Qb + ((size_t)(b * SS + qb0 + wave * 16 + laneM)) * DOUT + h * HD;
    qf[0].q[0] = *(const uint4*)(qp + koff0);
    qf[0].q[1] = *(const uint4*)(qp + koff0 + 16);
    qf[1].q[0] = *(const uint4*)(qp + 32 + koff0);
    qf[1].q[1] = *(const uint4*)(qp + 32 + koff0 + 16);
  }

  v8f o[4];
  float mrow[8], lrow[8];
#pragma unroll
  for (int d = 0; d < 4; ++d)
#pragma unroll
    for (int e = 0; e < 8; ++e) o[d][e] = 0.0f;
#pragma unroll
  for (int r = 0; r < 8; ++r) { mrow[r] = -3.0e38f; lrow[r] = 0.0f; }

  const int nblk = (qb0 >> 6) + 2;   // causal: k-blocks 0 .. qb0/64 + 1
  for (int kb = 0; kb < nblk; ++kb) {
    const int kbase = kb * 64;
    __syncthreads();
#pragma unroll
    for (int i = 0; i < 2; ++i) {    // stage 64x64 K (async DMA) and V^T
      int idx = tid + i * 256;
      int r = idx >> 3, c = (idx & 7) * 8;
      size_t goff = ((size_t)(b * SS + kbase + r)) * (NG * HD) + g * HD + c;
      async_ld_b128(&smK[r * KSTR + c], Kb + goff);
      union { uint4 q; unsigned short s[8]; } u;
      u.q = *(const uint4*)(Vb + goff);
#pragma unroll
      for (int j = 0; j < 8; ++j) smV[(c + j) * KSTR + r] = u.s[j];
    }
    wait_async0();
    __syncthreads();

    // scores: 16x64 = 4 C-tiles, contract over d (2 WMMA chunks)
    v8f st[4];
#pragma unroll
    for (int n = 0; n < 4; ++n) {
      int rr = n * 16 + laneM;
      FragAB k0, k1;
      k0.q[0] = *(const uint4*)(&smK[rr * KSTR + koff0]);
      k0.q[1] = *(const uint4*)(&smK[rr * KSTR + koff0 + 16]);
      k1.q[0] = *(const uint4*)(&smK[rr * KSTR + 32 + koff0]);
      k1.q[1] = *(const uint4*)(&smK[rr * KSTR + 32 + koff0 + 16]);
      v8f z;
#pragma unroll
      for (int e = 0; e < 8; ++e) z[e] = 0.0f;
      z = wmma_bf16(qf[0].v, k0.v, z);
      st[n] = wmma_bf16(qf[1].v, k1.v, z);
    }

    // scale + causal mask
    const int qrow0 = qb0 + wave * 16 + half8;
#pragma unroll
    for (int n = 0; n < 4; ++n)
#pragma unroll
      for (int r = 0; r < 8; ++r) {
        int kg = kbase + n * 16 + laneM;
        float s = st[n][r] * 0.125f;
        st[n][r] = (kg <= qrow0 + r) ? s : -3.0e38f;
      }

    // online softmax (row = cross-lane within each 16-lane half)
#pragma unroll
    for (int r = 0; r < 8; ++r) {
      float mc = fmaxf(fmaxf(st[0][r], st[1][r]), fmaxf(st[2][r], st[3][r]));
#pragma unroll
      for (int off = 1; off < 16; off <<= 1)
        mc = fmaxf(mc, __shfl_xor(mc, off, 32));
      float mnew = fmaxf(mrow[r], mc);
      float corr = __expf(mrow[r] - mnew);
      mrow[r] = mnew;
      float ps = 0.0f;
#pragma unroll
      for (int n = 0; n < 4; ++n) {
        float p = __expf(st[n][r] - mnew);
        st[n][r] = p;
        ps += p;
      }
#pragma unroll
      for (int off = 1; off < 16; off <<= 1) ps += __shfl_xor(ps, off, 32);
      lrow[r] = lrow[r] * corr + ps;
#pragma unroll
      for (int d = 0; d < 4; ++d) o[d][r] *= corr;
    }

    // restage P (C-layout -> A-layout) through per-wave LDS region
#pragma unroll
    for (int n = 0; n < 4; ++n)
#pragma unroll
      for (int r = 0; r < 8; ++r)
        smP[(wave * 16 + r + half8) * KSTR + n * 16 + laneM] = f2bf(st[n][r]);
    // same-wave DS ops are in order; no workgroup barrier needed here
    FragAB pf[2];
    {
      int pr = wave * 16 + laneM;
      pf[0].q[0] = *(const uint4*)(&smP[pr * KSTR + koff0]);
      pf[0].q[1] = *(const uint4*)(&smP[pr * KSTR + koff0 + 16]);
      pf[1].q[0] = *(const uint4*)(&smP[pr * KSTR + 32 + koff0]);
      pf[1].q[1] = *(const uint4*)(&smP[pr * KSTR + 32 + koff0 + 16]);
    }

    // O += P * V  (V^T staged in LDS, B-fragments read contiguous runs)
#pragma unroll
    for (int d = 0; d < 4; ++d) {
      int vr = d * 16 + laneM;
      FragAB v0, v1;
      v0.q[0] = *(const uint4*)(&smV[vr * KSTR + koff0]);
      v0.q[1] = *(const uint4*)(&smV[vr * KSTR + koff0 + 16]);
      v1.q[0] = *(const uint4*)(&smV[vr * KSTR + 32 + koff0]);
      v1.q[1] = *(const uint4*)(&smV[vr * KSTR + 32 + koff0 + 16]);
      o[d] = wmma_bf16(pf[0].v, v0.v, o[d]);
      o[d] = wmma_bf16(pf[1].v, v1.v, o[d]);
    }
  }

  // epilogue: O / l -> bf16 ctx[b, s, h*64 + d]
  float inv[8];
#pragma unroll
  for (int r = 0; r < 8; ++r) inv[r] = 1.0f / lrow[r];
#pragma unroll
  for (int d = 0; d < 4; ++d)
#pragma unroll
    for (int r = 0; r < 8; ++r) {
      int s = qb0 + wave * 16 + r + half8;
      int col = h * HD + d * 16 + laneM;
      Ctx[((size_t)(b * SS + s)) * DOUT + col] = f2bf(o[d][r] * inv[r]);
    }
}

// ---------------------------------------------------------------------------
// Host-side orchestration
// ---------------------------------------------------------------------------
extern "C" void kernel_launch(void* const* d_in, const int* in_sizes, int n_in,
                              void* d_out, int out_size, void* d_ws, size_t ws_size,
                              hipStream_t stream) {
  (void)in_sizes; (void)n_in; (void)out_size; (void)ws_size;
  const float* x  = (const float*)d_in[0];
  const float* Wq = (const float*)d_in[1];
  const float* Wk = (const float*)d_in[2];
  const float* Wv = (const float*)d_in[3];
  const float* Wo = (const float*)d_in[4];

  float* out    = (float*)d_out;                              // [B,S,2048]
  float* keys   = out + (size_t)BB * SS * DOUT;               // [B,G,S,64]
  float* values = keys + (size_t)BB * NG * SS * HD;           // [B,G,S,64]

  char* ws = (char*)d_ws;
  size_t off = 0;
  auto carve = [&](size_t bytes) { char* p = ws + off; off += (bytes + 255) & ~(size_t)255; return p; };
  unsigned short* xb  = (unsigned short*)carve((size_t)MALL * DIN * 2);
  unsigned short* wqb = (unsigned short*)carve((size_t)DOUT * DIN * 2);
  unsigned short* wkb = (unsigned short*)carve((size_t)NG * HD * DIN * 2);
  unsigned short* wvb = (unsigned short*)carve((size_t)NG * HD * DIN * 2);
  unsigned short* wob = (unsigned short*)carve((size_t)DIN * DOUT * 2);
  unsigned short* qb  = (unsigned short*)carve((size_t)MALL * DOUT * 2);
  unsigned short* kbf = (unsigned short*)carve((size_t)MALL * NG * HD * 2);
  unsigned short* vbf = (unsigned short*)carve((size_t)MALL * NG * HD * 2);
  unsigned short* ctx = (unsigned short*)carve((size_t)MALL * DOUT * 2);

  auto cvt = [&](const float* src, unsigned short* dst, int n) {
    cvt_f32_bf16<<<dim3((n / 4 + 255) / 256), dim3(256), 0, stream>>>(src, dst, n);
  };
  cvt(x,  xb,  MALL * DIN);
  cvt(Wq, wqb, DOUT * DIN);
  cvt(Wk, wkb, NG * HD * DIN);
  cvt(Wv, wvb, NG * HD * DIN);
  cvt(Wo, wob, DIN * DOUT);

  dim3 blk(256);
  // Q = x Wq^T  -> bf16
  gemm_bf16_wmma<<<dim3(DOUT / 128, MALL / 128), blk, 0, stream>>>(
      xb, wqb, qb, nullptr, MALL, DOUT, DIN, 0);
  // K = x Wk^T -> bf16 + f32 keys [b,g,s,d]
  gemm_bf16_wmma<<<dim3(NG * HD / 128, MALL / 128), blk, 0, stream>>>(
      xb, wkb, kbf, keys, MALL, NG * HD, DIN, 2);
  // V = x Wv^T -> bf16 + f32 values [b,g,s,d]
  gemm_bf16_wmma<<<dim3(NG * HD / 128, MALL / 128), blk, 0, stream>>>(
      xb, wvb, vbf, values, MALL, NG * HD, DIN, 2);
  // flash attention -> ctx bf16
  attn_wmma<<<dim3(SS / 128, BB * NH), blk, 0, stream>>>(qb, kbf, vbf, ctx);
  // out = ctx Wo^T -> f32 row-major
  gemm_bf16_wmma<<<dim3(DIN / 128, MALL / 128), blk, 0, stream>>>(
      ctx, wob, nullptr, out, MALL, DIN, DOUT, 1);
}
